// SparseAttention_21835613733530
// MI455X (gfx1250) — matrix operand order repaired and checked by reading, not compiled, not run on hardware
//
#include <hip/hip_runtime.h>
#include <hip/hip_bf16.h>

// ---------------------------------------------------------------------------
// Sparse attention (local window 128 + 16 global tokens, causal) on gfx1250.
// f16 WMMA (16x16x32) with f32 accumulation everywhere.
// ---------------------------------------------------------------------------

#define DM   1024
#define NH   16
#define DK   64
#define BB   2
#define SS   2048
#define MTOT (BB * SS)          // 4096 token rows
#define LWIN 128
#define GTOK 16

#if __has_builtin(__builtin_amdgcn_global_load_async_to_lds_b128)
#define HAVE_ASYNC_LDS 1
#else
#define HAVE_ASYNC_LDS 0
#endif

typedef __attribute__((ext_vector_type(16))) _Float16 v16h;
typedef __attribute__((ext_vector_type(8)))  _Float16 v8h;
typedef __attribute__((ext_vector_type(4)))  _Float16 v4h;
typedef __attribute__((ext_vector_type(8)))  float    v8f;
typedef __attribute__((ext_vector_type(4)))  int      v4i;

typedef __attribute__((address_space(1))) v4i* gv4i_p;   // global int4*
typedef __attribute__((address_space(3))) v4i* lv4i_p;   // LDS int4*

static __device__ __forceinline__ v16h combine16(v8h lo, v8h hi) {
    v16h r;
#pragma unroll
    for (int i = 0; i < 8; ++i) { r[i] = lo[i]; r[i + 8] = hi[i]; }
    return r;
}

static __device__ __forceinline__ v8f wmma_f16(v16h a, v16h b, v8f c) {
    // 8 args: (neg_a, A, neg_b, B, c_mod, C, reuse_a, reuse_b)
    return __builtin_amdgcn_wmma_f32_16x16x32_f16(
        false, a, false, b, (short)0, c, false, false);
}

// ---------------------------------------------------------------------------
// GEMM: Y[m][n] = sum_k X[m][k] * W[n][k] + bias[n]
// mode 0: X=f32 query -> qh  [B,H,S,DK] f16
// mode 1: X=f32 key   -> kh  [B,H,S,DK] f16
// mode 2: X=f32 value -> vT  [B,H,DK,S] f16 (transposed for attention B-frags)
// mode 3: X=f16 attnO -> out [B,S,DM]   f32  (A-tile staged via async->LDS)
// Workgroup: 256 threads = 8 waves; tile 128(M) x 128(N); k-step 32.
// ---------------------------------------------------------------------------
#define LDT 48   // LDS row stride in halves (96B, 16B-aligned, padded)

__global__ __launch_bounds__(256)
void proj_gemm(const float* __restrict__ Xf, const _Float16* __restrict__ Xh,
               const float* __restrict__ W, const float* __restrict__ bias,
               _Float16* __restrict__ outH, float* __restrict__ outF, int mode)
{
    __shared__ _Float16 As[128 * LDT];
    __shared__ _Float16 Bs[128 * LDT];

    const int m0   = blockIdx.x * 128;
    const int n0   = blockIdx.y * 128;
    const int t    = threadIdx.x;
    const int lane = t & 31;
    const int wave = t >> 5;          // 0..7
    const int wm   = wave & 3;        // 4 waves along M (32 rows each)
    const int wn   = wave >> 2;       // 2 waves along N (64 cols each)
    const int wrow = wm * 32;
    const int wcol = wn * 64;
    const int lrow  = lane & 15;
    const int lhalf = lane >> 4;

    v8f acc[2][4];
#pragma unroll
    for (int mt = 0; mt < 2; ++mt)
#pragma unroll
        for (int nt = 0; nt < 4; ++nt) {
            v8f z = {};
            acc[mt][nt] = z;
        }

    for (int k0 = 0; k0 < DM; k0 += 32) {
        // ---- stage A tile (128 x 32) into LDS as f16 ----
        if (mode != 3) {
            // phase 1: issue all global loads (clause-friendly)
            float4 fa[4];
#pragma unroll
            for (int i = 0; i < 4; ++i) {
                int v   = t + i * 256;        // float4 id, 0..1023
                int row = v >> 3;
                int c4  = v & 7;
                fa[i] = *(const float4*)(Xf + (size_t)(m0 + row) * DM + k0 + c4 * 4);
            }
            // phase 2: convert + LDS store
#pragma unroll
            for (int i = 0; i < 4; ++i) {
                int v   = t + i * 256;
                int row = v >> 3;
                int c4  = v & 7;
                v4h h; h[0] = (_Float16)fa[i].x; h[1] = (_Float16)fa[i].y;
                       h[2] = (_Float16)fa[i].z; h[3] = (_Float16)fa[i].w;
                *(v4h*)(&As[row * LDT + c4 * 4]) = h;
            }
        } else {
            // f16 source: DMA straight to LDS (ASYNCcnt path), no VGPR bounce
#pragma unroll
            for (int i = 0; i < 2; ++i) {
                int v   = t + i * 256;        // v8h id, 0..511
                int row = v >> 2;
                int c8  = v & 3;
                const _Float16* src = Xh + (size_t)(m0 + row) * DM + k0 + c8 * 8;
                _Float16*       dst = &As[row * LDT + c8 * 8];
#if HAVE_ASYNC_LDS
                __builtin_amdgcn_global_load_async_to_lds_b128(
                    (gv4i_p)src, (lv4i_p)dst, 0, 0);
#else
                *(v8h*)dst = *(const v8h*)src;
#endif
            }
        }
        // ---- stage B tile (weights, 128 n-rows x 32 k) into LDS as f16 ----
        {
            float4 fb[4];
#pragma unroll
            for (int i = 0; i < 4; ++i) {
                int v   = t + i * 256;
                int row = v >> 3;
                int c4  = v & 7;
                fb[i] = *(const float4*)(W + (size_t)(n0 + row) * DM + k0 + c4 * 4);
            }
#pragma unroll
            for (int i = 0; i < 4; ++i) {
                int v   = t + i * 256;
                int row = v >> 3;
                int c4  = v & 7;
                v4h h; h[0] = (_Float16)fb[i].x; h[1] = (_Float16)fb[i].y;
                       h[2] = (_Float16)fb[i].z; h[3] = (_Float16)fb[i].w;
                *(v4h*)(&Bs[row * LDT + c4 * 4]) = h;
            }
        }
        // prefetch next tiles into L2 while we compute
        if (k0 + 32 < DM) {
            if (mode != 3) __builtin_prefetch(Xf + (size_t)(m0 + (t >> 1)) * DM + k0 + 32, 0, 1);
            else           __builtin_prefetch(Xh + (size_t)(m0 + (t >> 1)) * DM + k0 + 32, 0, 1);
            __builtin_prefetch(W + (size_t)(n0 + (t >> 1)) * DM + k0 + 32, 0, 1);
        }
#if HAVE_ASYNC_LDS
        if (mode == 3) {
            // drain this wave's async LDS writes before the workgroup barrier
#if __has_builtin(__builtin_amdgcn_s_wait_asynccnt)
            __builtin_amdgcn_s_wait_asynccnt(0);
#else
            asm volatile("s_wait_asynccnt 0x0" ::: "memory");
#endif
        }
#endif
        __syncthreads();

        // ---- fragments ----
        v16h afrag[2], bfrag[4];
#pragma unroll
        for (int mt = 0; mt < 2; ++mt) {
            int r  = wrow + mt * 16 + lrow;
            int kb = lhalf * 8;           // A layout: K {0-7,16-23} / {8-15,24-31}
            v8h lo = *(const v8h*)(&As[r * LDT + kb]);
            v8h hi = *(const v8h*)(&As[r * LDT + kb + 16]);
            afrag[mt] = combine16(lo, hi);
        }
#pragma unroll
        for (int nt = 0; nt < 4; ++nt) {
            int n  = wcol + nt * 16 + lrow;
            int kb = lhalf * 8;           // B layout: K {0-7,16-23} / {8-15,24-31}
            v8h lo = *(const v8h*)(&Bs[n * LDT + kb]);
            v8h hi = *(const v8h*)(&Bs[n * LDT + kb + 16]);
            bfrag[nt] = combine16(lo, hi);
        }
#pragma unroll
        for (int mt = 0; mt < 2; ++mt)
#pragma unroll
            for (int nt = 0; nt < 4; ++nt)
                acc[mt][nt] = wmma_f16(afrag[mt], bfrag[nt], acc[mt][nt]);
        __syncthreads();
    }

    // ---- epilogue: bias + store ----
#pragma unroll
    for (int mt = 0; mt < 2; ++mt) {
#pragma unroll
        for (int nt = 0; nt < 4; ++nt) {
            int   n  = n0 + wcol + nt * 16 + lrow;
            float bv = bias[n];
#pragma unroll
            for (int r = 0; r < 8; ++r) {
                int   tok = m0 + wrow + mt * 16 + r + 8 * lhalf;  // C layout M
                float val = acc[mt][nt][r] + bv;
                if (mode == 3) {
                    outF[(size_t)tok * DM + n] = val;
                } else {
                    int bb = tok >> 11;        // /S
                    int s  = tok & (SS - 1);
                    int h  = n >> 6;           // /DK
                    int d  = n & (DK - 1);
                    if (mode == 2)             // vT: [B,H,DK,S]
                        outH[(((size_t)bb * NH + h) * DK + d) * SS + s] = (_Float16)val;
                    else                        // qh/kh: [B,H,S,DK]
                        outH[(((size_t)bb * NH + h) * SS + s) * DK + d] = (_Float16)val;
                }
            }
        }
    }
}

// ---------------------------------------------------------------------------
// Block-sparse flash attention. One wave = one 16-row query block of one head.
// Visits only key tiles intersecting the mask: tile 0 (global) + the 9 tiles
// covering [q0-128, q0]. Key tiles processed in pairs (WMMA K=32).
// ---------------------------------------------------------------------------
__global__ __launch_bounds__(128)
void attn_kernel(const _Float16* __restrict__ qh, const _Float16* __restrict__ kh,
                 const _Float16* __restrict__ vT, _Float16* __restrict__ oh)
{
    __shared__ _Float16 Plds[4][16 * 32];   // per-wave P staging (C->A layout)

    const int wave  = threadIdx.x >> 5;
    const int lane  = threadIdx.x & 31;
    const int blk   = blockIdx.x * 4 + wave;     // 0..4095
    const int qt    = blk & 127;                  // query tile within head
    const int bh    = blk >> 7;                   // b*NH + h
    const int q0    = qt * 16;
    const int lrow  = lane & 15;
    const int lhalf = lane >> 4;

    const _Float16* khh = kh + (size_t)bh * SS * DK;
    const _Float16* vTh = vT + (size_t)bh * DK * SS;

    // Q fragments: A-matrix 16x32, two d-chunks
    const _Float16* qbase = qh + ((size_t)bh * SS + q0 + lrow) * DK;
    v16h qa[2];
#pragma unroll
    for (int kk = 0; kk < 2; ++kk) {
        int kb = kk * 32 + lhalf * 8;
        v8h lo = *(const v8h*)(qbase + kb);
        v8h hi = *(const v8h*)(qbase + kb + 16);
        qa[kk] = combine16(lo, hi);
    }

    float mrun[8], lrun[8];
    v8f o[4];
#pragma unroll
    for (int r = 0; r < 8; ++r) { mrun[r] = -1e30f; lrun[r] = 0.0f; }
#pragma unroll
    for (int nt = 0; nt < 4; ++nt) { v8f z = {}; o[nt] = z; }

    const int jhi    = qt;
    const int jlo    = (q0 >= LWIN) ? (qt - LWIN / 16) : 0;
    const bool hasg  = (jlo > 0);
    const int ntiles = (jhi - jlo + 1) + (hasg ? 1 : 0);
    const int npairs = (ntiles + 1) >> 1;

    for (int p = 0; p < npairs; ++p) {
        const int  ia     = 2 * p, ib = 2 * p + 1;
        const int  jtA    = hasg ? (ia == 0 ? 0 : jlo + ia - 1) : (jlo + ia);
        const bool validB = (ib < ntiles);
        const int  jtB    = validB ? (hasg ? (jlo + ib - 1) : (jlo + ib)) : jtA;

        // ---- scores: S = Q K^T (K dim = d, two chunks of 32) ----
        v8f s0 = {}, s1 = {};
#pragma unroll
        for (int kk = 0; kk < 2; ++kk) {
            const _Float16* kpA = khh + ((size_t)(jtA * 16 + lrow)) * DK + kk * 32 + lhalf * 8;
            const _Float16* kpB = khh + ((size_t)(jtB * 16 + lrow)) * DK + kk * 32 + lhalf * 8;
            v16h kbA = combine16(*(const v8h*)kpA, *(const v8h*)(kpA + 16));
            v16h kbB = combine16(*(const v8h*)kpB, *(const v8h*)(kpB + 16));
            s0 = wmma_f16(qa[kk], kbA, s0);
            s1 = wmma_f16(qa[kk], kbB, s1);
        }

        // ---- mask + scale + row max ----
        const float sc = 0.125f;  // 1/sqrt(64)
        float rmax[8];
        const int jA = jtA * 16 + lrow;
        const int jB = jtB * 16 + lrow;
#pragma unroll
        for (int r = 0; r < 8; ++r) {
            int  i   = q0 + r + 8 * lhalf;
            bool okA = (jA <= i) && ((i - jA <= LWIN) || (jA < GTOK) || (i < GTOK));
            bool okB = validB &&
                       (jB <= i) && ((i - jB <= LWIN) || (jB < GTOK) || (i < GTOK));
            float a  = okA ? s0[r] * sc : -1e30f;
            float b2 = okB ? s1[r] * sc : -1e30f;
            s0[r] = a; s1[r] = b2;
            rmax[r] = fmaxf(a, b2);
        }
#pragma unroll
        for (int m = 1; m < 16; m <<= 1)
#pragma unroll
            for (int r = 0; r < 8; ++r)
                rmax[r] = fmaxf(rmax[r], __shfl_xor(rmax[r], m, 32));

        // ---- online softmax update ----
        float alpha[8], rs[8];
        v8f p0, p1;
#pragma unroll
        for (int r = 0; r < 8; ++r) {
            float nm = fmaxf(mrun[r], rmax[r]);
            alpha[r] = __expf(mrun[r] - nm);
            mrun[r]  = nm;
            p0[r] = __expf(s0[r] - nm);
            p1[r] = __expf(s1[r] - nm);
            rs[r] = p0[r] + p1[r];
        }
#pragma unroll
        for (int m = 1; m < 16; m <<= 1)
#pragma unroll
            for (int r = 0; r < 8; ++r)
                rs[r] += __shfl_xor(rs[r], m, 32);
#pragma unroll
        for (int r = 0; r < 8; ++r) lrun[r] = lrun[r] * alpha[r] + rs[r];
#pragma unroll
        for (int nt = 0; nt < 4; ++nt)
#pragma unroll
            for (int r = 0; r < 8; ++r) o[nt][r] *= alpha[r];

        // ---- P: C-layout f32 -> A-layout f16 via per-wave LDS ----
        _Float16* pl = &Plds[wave][0];
#pragma unroll
        for (int r = 0; r < 8; ++r) {
            int row = r + 8 * lhalf;
            pl[row * 32 + lrow]      = (_Float16)p0[r];
            pl[row * 32 + 16 + lrow] = (_Float16)p1[r];
        }
        asm volatile("s_wait_dscnt 0x0" ::: "memory");
        {
            int kb = lhalf * 16;
            v8h lo = *(const v8h*)(&pl[lrow * 32 + kb]);
            v8h hi = *(const v8h*)(&pl[lrow * 32 + kb + 8]);
            v16h pa = combine16(lo, hi);

            // ---- O += P * V : B-frags straight from transposed vT ----
            int keyb = (lhalf ? jtB : jtA) * 16;   // lanes<16: tileA keys, >=16: tileB
#pragma unroll
            for (int nt = 0; nt < 4; ++nt) {
                const _Float16* vp = vTh + ((size_t)(nt * 16 + lrow)) * SS + keyb;
                v16h vb = combine16(*(const v8h*)vp, *(const v8h*)(vp + 8));
                o[nt] = wmma_f16(pa, vb, o[nt]);
            }
        }
    }

    // ---- finalize: O /= l, store f16 to oh [B,S,DM] ----
    const int bb = bh >> 4;
    const int h  = bh & (NH - 1);
#pragma unroll
    for (int r = 0; r < 8; ++r) {
        float inv = 1.0f / lrun[r];
        int   tok = bb * SS + q0 + r + 8 * lhalf;
#pragma unroll
        for (int nt = 0; nt < 4; ++nt) {
            int col = h * DK + nt * 16 + lrow;
            oh[(size_t)tok * DM + col] = (_Float16)(o[nt][r] * inv);
        }
    }
}

// ---------------------------------------------------------------------------
extern "C" void kernel_launch(void* const* d_in, const int* in_sizes, int n_in,
                              void* d_out, int out_size, void* d_ws, size_t ws_size,
                              hipStream_t stream) {
    (void)in_sizes; (void)n_in; (void)out_size; (void)ws_size;
    const float* q  = (const float*)d_in[0];
    const float* k  = (const float*)d_in[1];
    const float* v  = (const float*)d_in[2];
    const float* wq = (const float*)d_in[3];
    const float* bq = (const float*)d_in[4];
    const float* wk = (const float*)d_in[5];
    const float* bk = (const float*)d_in[6];
    const float* wv = (const float*)d_in[7];
    const float* bv = (const float*)d_in[8];
    const float* wo = (const float*)d_in[9];
    const float* bo = (const float*)d_in[10];
    float* out = (float*)d_out;

    const size_t QKV = (size_t)MTOT * DM;   // 4,194,304 halves per buffer
    _Float16* ws = (_Float16*)d_ws;
    _Float16* qh = ws;
    _Float16* kh = ws + QKV;
    _Float16* vT = ws + 2 * QKV;
    _Float16* oh = ws + 3 * QKV;

    dim3 gg(MTOT / 128, DM / 128, 1);
    proj_gemm<<<gg, 256, 0, stream>>>(q, nullptr, wq, bq, qh, nullptr, 0);
    proj_gemm<<<gg, 256, 0, stream>>>(k, nullptr, wk, bk, kh, nullptr, 1);
    proj_gemm<<<gg, 256, 0, stream>>>(v, nullptr, wv, bv, vT, nullptr, 2);

    attn_kernel<<<(BB * NH * (SS / 16)) / 4, 128, 0, stream>>>(qh, kh, vT, oh);

    proj_gemm<<<gg, 256, 0, stream>>>(nullptr, oh, wo, bo, nullptr, out, 3);
}
